// TuckER_86466281603182
// MI455X (gfx1250) — matrix operand order repaired and checked by reading, not compiled, not run on hardware
//
#include <hip/hip_runtime.h>
#include <hip/hip_bf16.h>

typedef _Float16 half_t;
typedef __attribute__((ext_vector_type(8)))  _Float16 v8h;
typedef __attribute__((ext_vector_type(16))) _Float16 v16h;
typedef __attribute__((ext_vector_type(8)))  float    v8f;

#define D1 200
#define KDIM 80
#define N_REL 11
#define BATCH 1024
#define N_ENT 100000
#define KPAD 224            // 200 padded to 7*32 for wmma f16 k=32 steps
#define W1_LD 20100         // D1*(D1+1)/2
#define W2_LD 19900         // D1*(D1-1)/2

__constant__ int c_rel_perm[N_REL] = {8, 0, 3, 9, 4, 5, 6, 7, 1, 10, 2};

// ---------------------------------------------------------------------------
// K0: Rp[r, :] = R[REL_PERM[r], :]  with R assembled from R1|R2|R3 blocks
// ---------------------------------------------------------------------------
__global__ void build_Rp(const float* __restrict__ R1, const float* __restrict__ R2,
                         const float* __restrict__ R3, float* __restrict__ Rp) {
    for (int t = threadIdx.x; t < N_REL * KDIM; t += blockDim.x) {
        int r = t / KDIM, i = t % KDIM;
        int src = c_rel_perm[r];
        float v;
        if (src < 3)       v = (i < 30) ? R1[src * 30 + i] : 0.f;
        else if (src < 8)  v = (i >= 30 && i < 60) ? R2[(src - 3) * 30 + (i - 30)] : 0.f;
        else               v = R3[(src - 8) * KDIM + i];
        Rp[t] = v;
    }
}

// ---------------------------------------------------------------------------
// K1: W_rel[r,j,k] = sum_i Rp[r,i] * W[i,j,k] (W built on the fly from W1/W2/W3)
//   sym slice s:  W1[s, T(min(j,k),max(j,k))],  T(i,j)=i*D1 - i(i-1)/2 + (j-i)
//   asym slice a: sign * W2[a, U(lo,hi)],       U(i,j)=i*D1 - i(i+1)/2 + (j-i-1)
// ---------------------------------------------------------------------------
__global__ void build_wrel(const float* __restrict__ Rp, const float* __restrict__ W1,
                           const float* __restrict__ W2, const float* __restrict__ W3,
                           float* __restrict__ W_rel) {
    __shared__ float rloc[KDIM];
    int r = blockIdx.y;
    if (threadIdx.x < KDIM) rloc[threadIdx.x] = Rp[r * KDIM + threadIdx.x];
    __syncthreads();
    int idx = blockIdx.x * blockDim.x + threadIdx.x;
    if (idx >= D1 * D1) return;
    int j = idx / D1, k = idx % D1;
    int lo = j < k ? j : k;
    int hi = j < k ? k : j;
    int tcol = lo * D1 - (lo * (lo - 1)) / 2 + (hi - lo);
    float acc = 0.f;
#pragma unroll 6
    for (int s = 0; s < 30; ++s) acc = fmaf(rloc[s], W1[s * W1_LD + tcol], acc);
    if (j != k) {
        int ucol = lo * D1 - (lo * (lo + 1)) / 2 + (hi - lo - 1);
        float sgn = (k > j) ? 1.f : -1.f;
#pragma unroll 6
        for (int a = 0; a < 30; ++a) acc = fmaf(sgn * rloc[30 + a], W2[a * W2_LD + ucol], acc);
    }
#pragma unroll 5
    for (int o = 0; o < 20; ++o) acc = fmaf(rloc[60 + o], W3[o * D1 * D1 + idx], acc);
    W_rel[r * D1 * D1 + idx] = acc;
}

// ---------------------------------------------------------------------------
// K2/K4: training-mode batchnorm statistics (biased var) over 1024 rows,
// one feature column per block. gidx==nullptr -> identity row indexing.
// ---------------------------------------------------------------------------
__global__ void bn_stats(const float* __restrict__ X, const int* __restrict__ gidx,
                         int ld, float* __restrict__ mean, float* __restrict__ rstd) {
    __shared__ float s1[256];
    __shared__ float s2[256];
    int col = blockIdx.x;
    int t = threadIdx.x;
    float a1 = 0.f, a2 = 0.f;
    for (int b = t; b < BATCH; b += 256) {
        int row = gidx ? gidx[b] : b;
        float v = X[(size_t)row * ld + col];
        a1 += v;
        a2 += v * v;
    }
    s1[t] = a1; s2[t] = a2;
    __syncthreads();
    for (int s = 128; s > 0; s >>= 1) {
        if (t < s) { s1[t] += s1[t + s]; s2[t] += s2[t + s]; }
        __syncthreads();
    }
    if (t == 0) {
        float m = s1[0] * (1.f / BATCH);
        float v = s2[0] * (1.f / BATCH) - m * m;
        mean[col] = m;
        rstd[col] = rsqrtf(v + 1e-5f);
    }
}

// ---------------------------------------------------------------------------
// K3: y[b,k] = sum_j bn0(E[e1[b],j]) * W_rel[r_idx[b], j, k]
// ---------------------------------------------------------------------------
__global__ void bn_matvec(const float* __restrict__ E, const int* __restrict__ e1_idx,
                          const int* __restrict__ r_idx,
                          const float* __restrict__ m0, const float* __restrict__ rs0,
                          const float* __restrict__ g0, const float* __restrict__ b0,
                          const float* __restrict__ W_rel, float* __restrict__ y) {
    __shared__ float x0[D1];
    int b = blockIdx.x;
    int t = threadIdx.x;
    const float* Erow = E + (size_t)e1_idx[b] * D1;
    if (t < D1) x0[t] = (Erow[t] - m0[t]) * rs0[t] * g0[t] + b0[t];
    __syncthreads();
    if (t < D1) {
        const float* W = W_rel + (size_t)r_idx[b] * (D1 * D1);
        float acc = 0.f;
        for (int j = 0; j < D1; ++j) acc = fmaf(x0[j], W[j * D1 + t], acc);
        y[b * D1 + t] = acc;
    }
}

// ---------------------------------------------------------------------------
// K5a: Ah[b,k] = f16( bn1(y[b,k]) ), zero-padded to KPAD
// ---------------------------------------------------------------------------
__global__ void conv_x(const float* __restrict__ y,
                       const float* __restrict__ m1, const float* __restrict__ rs1,
                       const float* __restrict__ g1, const float* __restrict__ b1,
                       half_t* __restrict__ Ah) {
    int i = blockIdx.x * blockDim.x + threadIdx.x;
    if (i >= BATCH * KPAD) return;
    int b = i / KPAD, k = i % KPAD;
    float v = 0.f;
    if (k < D1) v = (y[b * D1 + k] - m1[k]) * rs1[k] * g1[k] + b1[k];
    Ah[i] = (half_t)v;
}

// K5b: Eh[n,k] = f16(E[n,k]), zero-padded to KPAD
__global__ void conv_E(const float* __restrict__ E, half_t* __restrict__ Eh) {
    int i = blockIdx.x * blockDim.x + threadIdx.x;
    if (i >= N_ENT * KPAD) return;
    int n = i / KPAD, k = i % KPAD;
    Eh[i] = (half_t)((k < D1) ? E[(size_t)n * D1 + k] : 0.f);
}

// ---------------------------------------------------------------------------
// K6: out[b,n] = sigmoid( sum_k Ah[b,k] * Eh[n,k] ) via v_wmma_f32_16x16x32_f16
// One 16x16 tile per wave; 8 waves/block cover 8 N-tiles.
// A layout (16-bit 16x32): lanes 0-15 -> M=lane, halves 0-7 = K0..7, 8-15 = K16..23;
//                          lanes 16-31 -> K shifted by 8.
// B layout (16-bit 32x16): lane -> N, same K order as A (halves 0-7 = K0..7, 8-15 = K16..23; lanes 16-31 -> K shifted by 8).
// ---------------------------------------------------------------------------
__global__ void __launch_bounds__(256)
gemm_sigmoid(const half_t* __restrict__ Ah, const half_t* __restrict__ Eh,
             float* __restrict__ out) {
    const int n_tiles = N_ENT / 16;   // 6250
    int wave = threadIdx.x >> 5;
    int lane = threadIdx.x & 31;
    int n_tile = blockIdx.x * 8 + wave;
    if (n_tile >= n_tiles) return;    // wave-uniform: surviving waves keep EXEC all-1s
    int m_tile = blockIdx.y;
    int ll = lane & 15;
    bool hiL = lane >= 16;

    const half_t* arow = Ah + (size_t)(m_tile * 16 + ll) * KPAD;
    const half_t* brow = Eh + (size_t)(n_tile * 16 + ll) * KPAD;

    v8f c = {};
#pragma unroll
    for (int kk = 0; kk < 7; ++kk) {
        int k0 = kk * 32;
        v8h a_lo = *(const v8h*)(arow + k0 + (hiL ? 8 : 0));
        v8h a_hi = *(const v8h*)(arow + k0 + 16 + (hiL ? 8 : 0));
        v8h b_lo = *(const v8h*)(brow + k0 + (hiL ? 8 : 0));
        v8h b_hi = *(const v8h*)(brow + k0 + 16 + (hiL ? 8 : 0));
        v16h a, bm;
#pragma unroll
        for (int i = 0; i < 8; ++i) {
            a[i] = a_lo[i];  a[i + 8] = a_hi[i];
            bm[i] = b_lo[i]; bm[i + 8] = b_hi[i];
        }
        c = __builtin_amdgcn_wmma_f32_16x16x32_f16(false, a, false, bm,
                                                   (short)0, c, false, false);
    }

    int col = n_tile * 16 + ll;
    int mbase = m_tile * 16 + (hiL ? 8 : 0);
#pragma unroll
    for (int v = 0; v < 8; ++v) {
        float s = 1.0f / (1.0f + __expf(-c[v]));
        // 410 MB of output: stream past L2 so Eh stays resident
        __builtin_nontemporal_store(s, &out[(size_t)(mbase + v) * N_ENT + col]);
    }
}

// ---------------------------------------------------------------------------
extern "C" void kernel_launch(void* const* d_in, const int* in_sizes, int n_in,
                              void* d_out, int out_size, void* d_ws, size_t ws_size,
                              hipStream_t stream) {
    const float* E   = (const float*)d_in[0];
    const float* R1  = (const float*)d_in[1];
    const float* R2  = (const float*)d_in[2];
    const float* R3  = (const float*)d_in[3];
    const float* W1  = (const float*)d_in[4];
    const float* W2  = (const float*)d_in[5];
    const float* W3  = (const float*)d_in[6];
    const float* g0  = (const float*)d_in[7];
    const float* b0  = (const float*)d_in[8];
    const float* g1  = (const float*)d_in[9];
    const float* b1  = (const float*)d_in[10];
    const int* e1_idx = (const int*)d_in[11];
    const int* r_idx  = (const int*)d_in[12];
    float* out = (float*)d_out;

    // workspace carve-out (256B aligned)
    char* ws = (char*)d_ws;
    size_t off = 0;
    auto alloc = [&](size_t bytes) {
        char* p = ws + off;
        off = (off + bytes + 255) & ~(size_t)255;
        return p;
    };
    float*  Rp    = (float*)alloc(N_REL * KDIM * sizeof(float));
    float*  W_rel = (float*)alloc((size_t)N_REL * D1 * D1 * sizeof(float));
    float*  m0    = (float*)alloc(D1 * sizeof(float));
    float*  rs0   = (float*)alloc(D1 * sizeof(float));
    float*  m1    = (float*)alloc(D1 * sizeof(float));
    float*  rs1   = (float*)alloc(D1 * sizeof(float));
    float*  yb    = (float*)alloc((size_t)BATCH * D1 * sizeof(float));
    half_t* Ah    = (half_t*)alloc((size_t)BATCH * KPAD * sizeof(half_t));
    half_t* Eh    = (half_t*)alloc((size_t)N_ENT * KPAD * sizeof(half_t));
    (void)ws_size; (void)in_sizes; (void)n_in; (void)out_size;

    build_Rp<<<1, 128, 0, stream>>>(R1, R2, R3, Rp);

    dim3 g1d((D1 * D1 + 255) / 256, N_REL);
    build_wrel<<<g1d, 256, 0, stream>>>(Rp, W1, W2, W3, W_rel);

    bn_stats<<<D1, 256, 0, stream>>>(E, e1_idx, D1, m0, rs0);

    bn_matvec<<<BATCH, 256, 0, stream>>>(E, e1_idx, r_idx, m0, rs0, g0, b0, W_rel, yb);

    bn_stats<<<D1, 256, 0, stream>>>(yb, nullptr, D1, m1, rs1);

    conv_x<<<(BATCH * KPAD + 255) / 256, 256, 0, stream>>>(yb, m1, rs1, g1, b1, Ah);
    conv_E<<<(N_ENT * KPAD + 255) / 256, 256, 0, stream>>>(E, Eh);

    dim3 gg((N_ENT / 16 + 7) / 8, BATCH / 16);  // (782, 64), 8 waves/block
    gemm_sigmoid<<<gg, 256, 0, stream>>>(Ah, Eh, out);
}
